// Transformer_Based_Model_9294309229337
// MI455X (gfx1250) — compile-verified, matrix operand order checked by reading, not run on hardware
//
#include <hip/hip_runtime.h>
#include <hip/hip_bf16.h>
#include <math.h>

// Problem constants (match reference)
#define BB   2
#define SS   2048
#define DDm  1024
#define HHn  16
#define DHH  64
#define KTOP 1433   // int(S * 0.7)

typedef __bf16 bf16_t;
typedef __attribute__((ext_vector_type(16))) __bf16 v16bf;
typedef __attribute__((ext_vector_type(8)))  __bf16 v8bf;   // one 16B run
typedef __attribute__((ext_vector_type(4)))  __bf16 v4bf;
typedef __attribute__((ext_vector_type(8)))  float  v8f;
typedef __attribute__((ext_vector_type(4)))  unsigned v4u;
typedef __attribute__((ext_vector_type(8)))  int      v8i;
typedef __attribute__((ext_vector_type(4)))  int      v4i;

// A/B 16-bit fragments: per lane, elements 0..7 = K run at half*8,
// elements 8..15 = K run at 16+half*8 (CDNA5 ISA 7.12.2).
union FragBF { v16bf v; v8bf h[2]; };

// ---- gfx1250 async global->LDS copy (ASYNCcnt-tracked) ----
typedef __attribute__((address_space(3))) char* lds_ptr_t;
__device__ __forceinline__ unsigned lds_off(void* p) {
    return (unsigned)(size_t)(lds_ptr_t)p;     // generic -> LDS offset
}
__device__ __forceinline__ void async_cp16(unsigned ldsoff, const bf16_t* g) {
    asm volatile("global_load_async_to_lds_b128 %0, %1, off"
                 :: "v"(ldsoff), "v"(g) : "memory");
}
__device__ __forceinline__ void wait_async0() {
    asm volatile("s_wait_asynccnt 0x0" ::: "memory");
}

// ---- gfx1250 Tensor Data Mover: 2D tile load, LDS row padding 64B+16B ----
// D# per CDNA5 ISA. data_size=2 bytes.
// pad_interval=3 (pad after 16 DWORDs = 64B row), pad_amount=3 (4 DWORDs=16B)
// -> LDS pitch 80B, matching the 40-element fragment layout.
// This toolchain exposes the 6-arg builtin:
//   (uint32x4 g0, int32x8 g1, int32x4, int32x4, int32x8, i32 cpol)
__device__ __forceinline__ void tdm_load_2d(unsigned lds_addr, const void* gptr,
                                            unsigned tensor_d0, unsigned tensor_d1,
                                            unsigned tile_d0, unsigned tile_d1,
                                            unsigned d0_stride) {
    unsigned long long ga = (unsigned long long)(size_t)gptr;
    v4u g0;
    g0[0] = 1u;                                            // count=1 (valid), user
    g0[1] = lds_addr;                                      // lds byte address
    g0[2] = (unsigned)(ga & 0xFFFFFFFFu);                  // global_addr[31:0]
    g0[3] = (unsigned)((ga >> 32) & 0x01FFFFFFu) | (2u << 30); // addr[56:32]|type=2
    v8i g1;
    g1[0] = (int)((1u << 16) |                             // data_size=1 -> 2B
                  (1u << 20) |                             // pad_enable
                  (3u << 22) |                             // pad_interval: 16 DW
                  (3u << 25));                             // pad_amount: 4 DW
    g1[1] = (int)((tensor_d0 & 0xFFFFu) << 16);            // dim0[15:0] (atomic=0)
    g1[2] = (int)(((tensor_d0 >> 16) & 0xFFFFu) | ((tensor_d1 & 0xFFFFu) << 16));
    g1[3] = (int)(((tensor_d1 >> 16) & 0xFFFFu) | ((tile_d0 & 0xFFFFu) << 16));
    g1[4] = (int)(tile_d1 & 0xFFFFu);                      // tile_dim1, tile_dim2=0
    g1[5] = (int)d0_stride;                                // dim0_stride[31:0]
    g1[6] = 0;                                             // stride hi / dim1_stride
    g1[7] = 0;
    v4i z4 = { 0, 0, 0, 0 };
    v8i z8 = { 0, 0, 0, 0, 0, 0, 0, 0 };
    __builtin_amdgcn_tensor_load_to_lds(g0, g1, z4, z4, z8, 0);
}

// Monotonic float <-> sortable-uint key mapping
__device__ __forceinline__ unsigned float_to_key(float f) {
    unsigned u = __float_as_uint(f);
    return (u & 0x80000000u) ? ~u : (u | 0x80000000u);
}
__device__ __forceinline__ float key_to_float(unsigned k) {
    unsigned u = (k & 0x80000000u) ? (k & 0x7FFFFFFFu) : ~k;
    return __uint_as_float(u);
}

// =====================================================================
// Weight prep: fp32 [K,N] -> bf16 [N,K] (transposed once, LDS-tiled)
// =====================================================================
__global__ __launch_bounds__(256)
void cvt_transpose_kernel(const float* __restrict__ in, bf16_t* __restrict__ out,
                          int Kd, int N) {
    __shared__ float tile[32][33];
    const int k0 = blockIdx.x * 32, n0 = blockIdx.y * 32;
    const int tx = threadIdx.x & 31, ty = threadIdx.x >> 5;   // 32 x 8
    for (int j = 0; j < 32; j += 8)
        tile[ty + j][tx] = in[(size_t)(k0 + ty + j) * N + n0 + tx];
    __syncthreads();
    for (int j = 0; j < 32; j += 8)
        out[(size_t)(n0 + ty + j) * Kd + k0 + tx] = (bf16_t)tile[tx][ty + j];
}

// =====================================================================
// LayerNorm: one row (D=1024) per 256-thread block.
// =====================================================================
__global__ __launch_bounds__(256)
void ln_kernel(const float* __restrict__ x, const float* __restrict__ g,
               const float* __restrict__ bta, float* __restrict__ out_f32,
               bf16_t* __restrict__ out_bf) {
    __shared__ float s1[256], s2[256];
    const int row = blockIdx.x;
    const int tid = threadIdx.x;
    const float* xr = x + (size_t)row * DDm;
    float4 v4 = *(const float4*)&xr[tid * 4];
    float vals[4] = { v4.x, v4.y, v4.z, v4.w };
    float a = v4.x + v4.y + v4.z + v4.w;
    float b = v4.x * v4.x + v4.y * v4.y + v4.z * v4.z + v4.w * v4.w;
    s1[tid] = a; s2[tid] = b;
    __syncthreads();
    for (int off = 128; off > 0; off >>= 1) {
        if (tid < off) { s1[tid] += s1[tid + off]; s2[tid] += s2[tid + off]; }
        __syncthreads();
    }
    const float mu  = s1[0] * (1.0f / DDm);
    const float var = s2[0] * (1.0f / DDm) - mu * mu;
    const float rs  = rsqrtf(var + 1e-5f);
    float4 of; v4bf ob;
    for (int j = 0; j < 4; ++j) {
        int d = tid * 4 + j;
        float v = (vals[j] - mu) * rs * g[d] + bta[d];
        (&of.x)[j] = v;
        ob[j] = (bf16_t)v;
    }
    if (out_f32) *(float4*)&out_f32[(size_t)row * DDm + tid * 4] = of;
    *(v4bf*)&out_bf[(size_t)row * DDm + tid * 4] = ob;
}

// =====================================================================
// bf16 WMMA GEMM:  out = A[M,K] @ WT[N,K]^T + bias, fused epilogues.
// Block tile 256x64, 8 waves, each wave 32x64 (2 A-frags, 8 accums).
// A tile (256x32) staged by the Tensor Data Mover (wave 0 issues the
// descriptor, TENSORcnt-tracked, LDS padding by the DMA itself);
// B tile (64x32) staged by per-thread global_load_async_to_lds_b128
// (ASYNCcnt-tracked). Double buffered; fragments are 2x ds_load_b128.
// mode 0: out_bf = (acc+bias)*scale, head-permuted [B,H,S,DH]   (Q,K)
// mode 3: out_bf = acc+bias, transposed [B,H,DH,S]              (V)
// mode 1: out_f32 = acc + bias + resid                          (Wo/W2)
// mode 2: out_bf  = exact GELU(acc + bias)                      (W1)
// =====================================================================
__global__ __launch_bounds__(256)
void gemm_bf16_kernel(const bf16_t* __restrict__ A, const bf16_t* __restrict__ WT,
                      const float* __restrict__ bias, int M, int N, int Kd,
                      int mode, float scale, const float* __restrict__ resid,
                      float* __restrict__ out_f32, bf16_t* __restrict__ out_bf) {
    constexpr int LDA  = 40;   // 32 + pad; pitch 80B = TDM pad setting
    constexpr int LDBT = 40;
    __shared__ bf16_t As[2][256 * LDA];
    __shared__ bf16_t BsT[2][64 * LDBT];

    const int tid  = threadIdx.x;
    const int lane = tid & 31;
    const int wv   = tid >> 5;
    const int half = lane >> 4;
    const int m0   = blockIdx.y * 256;
    const int n0   = blockIdx.x * 64;

    // B staging: 64x32 = 256 16B chunks, 1 per thread
    const int br = tid >> 2, bk = (tid & 3) * 8;

    auto stage = [&](int kb, int buf) {
        if (wv == 0) {   // TDM: whole 256x32 A tile in one descriptor
            tdm_load_2d(lds_off(&As[buf][0]),
                        A + (size_t)m0 * Kd + kb,
                        (unsigned)Kd, (unsigned)M,
                        32u, 256u, (unsigned)Kd);
        }
        async_cp16(lds_off(&BsT[buf][br * LDBT + bk]),
                   &WT[(size_t)(n0 + br) * Kd + kb + bk]);
    };

    v8f c[2][4] = {};

    stage(0, 0);
    wait_async0();
    __builtin_amdgcn_s_wait_tensorcnt(0);
    __syncthreads();

    int buf = 0;
    for (int kb = 0; kb < Kd; kb += 32) {
        if (kb + 32 < Kd) stage(kb + 32, buf ^ 1);   // overlap next tile

        FragBF a0, a1;
        {
            const bf16_t* ar = &As[buf][(wv * 32 + (lane & 15)) * LDA];
            a0.h[0] = *(const v8bf*)(ar + half * 8);
            a0.h[1] = *(const v8bf*)(ar + 16 + half * 8);
            ar += 16 * LDA;
            a1.h[0] = *(const v8bf*)(ar + half * 8);
            a1.h[1] = *(const v8bf*)(ar + 16 + half * 8);
        }
        for (int nt = 0; nt < 4; ++nt) {
            FragBF b;
            const bf16_t* brp = &BsT[buf][(nt * 16 + (lane & 15)) * LDBT];
            b.h[0] = *(const v8bf*)(brp + half * 8);
            b.h[1] = *(const v8bf*)(brp + 16 + half * 8);
            c[0][nt] = __builtin_amdgcn_wmma_f32_16x16x32_bf16(
                false, a0.v, false, b.v, (short)0, c[0][nt], false, false);
            c[1][nt] = __builtin_amdgcn_wmma_f32_16x16x32_bf16(
                false, a1.v, false, b.v, (short)0, c[1][nt], false, false);
        }
        wait_async0();
        __builtin_amdgcn_s_wait_tensorcnt(0);
        __syncthreads();
        buf ^= 1;
    }

    // epilogue (C layout: VGPR i -> M = i + half*8, N = lane&15 + 16*nt)
    for (int rt = 0; rt < 2; ++rt) {
        for (int nt = 0; nt < 4; ++nt) {
            for (int i = 0; i < 8; ++i) {
                int m    = i + half * 8;
                int n    = nt * 16 + (lane & 15);
                int grow = m0 + wv * 32 + rt * 16 + m;
                int gcol = n0 + n;
                float v  = c[rt][nt][i] + bias[gcol];
                if (mode == 0) {
                    v *= scale;
                    int bq = grow / SS, s = grow % SS;
                    int h  = gcol >> 6, dh = gcol & 63;
                    out_bf[(((size_t)bq * HHn + h) * SS + s) * DHH + dh] = (bf16_t)v;
                } else if (mode == 3) {
                    int bq = grow / SS, s = grow % SS;
                    int h  = gcol >> 6, dh = gcol & 63;
                    out_bf[(((size_t)bq * HHn + h) * DHH + dh) * SS + s] = (bf16_t)v;
                } else if (mode == 1) {
                    size_t o = (size_t)grow * N + gcol;
                    out_f32[o] = v + resid[o];
                } else {
                    float gl = 0.5f * v * (1.0f + erff(v * 0.70710678118654752f));
                    out_bf[(size_t)grow * N + gcol] = (bf16_t)gl;
                }
            }
        }
    }
}

// =====================================================================
// Fused top-k attention per (b, h, 16-query-row tile). 256 thr = 8 waves.
// 16x2048 fp32 score strip in LDS (128 KB). Q/K [B,H,S,DH], V [B,H,DH,S].
// =====================================================================
__global__ __launch_bounds__(256)
void attn_topk_kernel(const bf16_t* __restrict__ Q, const bf16_t* __restrict__ Kt,
                      const bf16_t* __restrict__ Vt, bf16_t* __restrict__ ctx) {
    __shared__ float sc[16 * 2048];   // scores -> probs -> AV partials

    const int tid  = threadIdx.x;
    const int lane = tid & 31;
    const int wv   = tid >> 5;
    const int half = lane >> 4;
    const int bh   = blockIdx.x >> 7;     // / (S/16)
    const int q0   = (blockIdx.x & 127) * 16;

    const bf16_t* qb  = Q  + (size_t)bh * SS * DHH;
    const bf16_t* kpb = Kt + (size_t)bh * SS * DHH;
    const bf16_t* vtb = Vt + (size_t)bh * DHH * SS;

    // Q fragments (dh 0..31 and 32..63): 2x b128 each
    FragBF qa0, qa1;
    {
        const bf16_t* qr = qb + (size_t)(q0 + (lane & 15)) * DHH;
        qa0.h[0] = *(const v8bf*)(qr + half * 8);
        qa0.h[1] = *(const v8bf*)(qr + 16 + half * 8);
        qa1.h[0] = *(const v8bf*)(qr + 32 + half * 8);
        qa1.h[1] = *(const v8bf*)(qr + 48 + half * 8);
    }

    // ---- Phase 1: scores S[16 x 2048], 16-key tiles round-robin ----
    for (int t = wv; t < 128; t += 8) {
        int kt0 = t * 16;
        int n   = lane & 15;
        const bf16_t* kr = kpb + (size_t)(kt0 + n) * DHH;
        if (t + 8 < 128)   // prefetch next tile's K rows (global_prefetch_b8)
            __builtin_prefetch(kpb + (size_t)(kt0 + 128 + n) * DHH, 0, 1);
        FragBF b0, b1;
        b0.h[0] = *(const v8bf*)(kr + half * 8);
        b0.h[1] = *(const v8bf*)(kr + 16 + half * 8);
        b1.h[0] = *(const v8bf*)(kr + 32 + half * 8);
        b1.h[1] = *(const v8bf*)(kr + 48 + half * 8);
        v8f cc = {};
        cc = __builtin_amdgcn_wmma_f32_16x16x32_bf16(false, qa0.v, false, b0.v, (short)0, cc, false, false);
        cc = __builtin_amdgcn_wmma_f32_16x16x32_bf16(false, qa1.v, false, b1.v, (short)0, cc, false, false);
        for (int i = 0; i < 8; ++i)
            sc[(i + half * 8) * 2048 + kt0 + n] = cc[i];
    }
    __syncthreads();

    // ---- Selection + softmax; wave wv owns rows wv and wv+8.
    for (int rr = 0; rr < 2; ++rr) {
        const int r = wv + rr * 8;
        float* prow = sc + r * 2048 + lane * 64;

        unsigned kreg[64];
        unsigned mk = 0;
        for (int j4 = 0; j4 < 16; ++j4) {
            float4 f = *(const float4*)&prow[j4 * 4];
            unsigned k0 = float_to_key(f.x), k1 = float_to_key(f.y);
            unsigned k2 = float_to_key(f.z), k3 = float_to_key(f.w);
            kreg[j4 * 4 + 0] = k0; kreg[j4 * 4 + 1] = k1;
            kreg[j4 * 4 + 2] = k2; kreg[j4 * 4 + 3] = k3;
            unsigned a = k0 > k1 ? k0 : k1, b = k2 > k3 ? k2 : k3;
            unsigned m2 = a > b ? a : b;
            if (m2 > mk) mk = m2;
        }
        for (int off = 16; off; off >>= 1) {
            unsigned o = (unsigned)__shfl_xor((int)mk, off, 32);
            if (o > mk) mk = o;
        }
        // exact K-th largest via 32-step bit-prefix search (registers only)
        unsigned prefix = 0;
        for (int bbit = 31; bbit >= 0; --bbit) {
            unsigned cand = prefix | (1u << bbit);
            int cnt = 0;
            for (int j = 0; j < 64; ++j) cnt += (kreg[j] >= cand) ? 1 : 0;
            for (int off = 16; off; off >>= 1) cnt += __shfl_xor(cnt, off, 32);
            if (cnt >= KTOP) prefix = cand;
        }
        const float thrf = key_to_float(prefix);
        const float mx   = key_to_float(mk);

        float sum = 0.f;
        for (int j4 = 0; j4 < 16; ++j4) {
            float4 e;
            for (int j = 0; j < 4; ++j) {
                float f = key_to_float(kreg[j4 * 4 + j]);
                (&e.x)[j] = (f >= thrf) ? __expf(f - mx) : 0.f;
            }
            sum += e.x + e.y + e.z + e.w;
            *(float4*)&prow[j4 * 4] = e;
        }
        for (int off = 16; off; off >>= 1) sum += __shfl_xor(sum, off, 32);
        const float inv = 1.0f / sum;
        for (int j4 = 0; j4 < 16; ++j4) {
            float4 e = *(const float4*)&prow[j4 * 4];
            e.x *= inv; e.y *= inv; e.z *= inv; e.w *= inv;
            *(float4*)&prow[j4 * 4] = e;
        }
    }
    __syncthreads();

    // ---- Phase 4: ctx(16x64) = P(16x2048) . V; keys split across waves
    v8f c[4] = {};
    for (int k0 = wv * 256; k0 < wv * 256 + 256; k0 += 32) {
        FragBF a;
        {
            const float* pr = sc + (lane & 15) * 2048 + k0;
            v8f flo = *(const v8f*)(pr + half * 8);
            v8f fhi = *(const v8f*)(pr + 16 + half * 8);
            for (int e = 0; e < 8; ++e) {
                a.h[0][e] = (bf16_t)flo[e];
                a.h[1][e] = (bf16_t)fhi[e];
            }
        }
        for (int nt = 0; nt < 4; ++nt) {
            const bf16_t* vr = vtb + (size_t)(nt * 16 + (lane & 15)) * SS + k0;
            FragBF b;
            b.h[0] = *(const v8bf*)(vr + half * 8);
            b.h[1] = *(const v8bf*)(vr + 16 + half * 8);
            c[nt] = __builtin_amdgcn_wmma_f32_16x16x32_bf16(
                false, a.v, false, b.v, (short)0, c[nt], false, false);
        }
    }
    __syncthreads();                      // probs dead: reuse sc for partials
    float* part = sc;                     // [8 waves][16][64]
    for (int nt = 0; nt < 4; ++nt)
        for (int i = 0; i < 8; ++i)
            part[wv * 1024 + (i + half * 8) * 64 + nt * 16 + (lane & 15)] = c[nt][i];
    __syncthreads();
    const int b_ = bh / HHn, h_ = bh % HHn;
    for (int i = tid; i < 1024; i += 256) {
        float s = 0.f;
        for (int w2 = 0; w2 < 8; ++w2) s += part[w2 * 1024 + i];
        int m = i >> 6, n = i & 63;
        ctx[((size_t)b_ * SS + q0 + m) * DDm + h_ * DHH + n] = (bf16_t)s;
    }
}

// =====================================================================
// Host-side orchestration
// =====================================================================
extern "C" void kernel_launch(void* const* d_in, const int* in_sizes, int n_in,
                              void* d_out, int out_size, void* d_ws, size_t ws_size,
                              hipStream_t stream) {
    const float* x     = (const float*)d_in[0];
    const float* Wq    = (const float*)d_in[1];
    const float* bq    = (const float*)d_in[2];
    const float* Wk    = (const float*)d_in[3];
    const float* bk    = (const float*)d_in[4];
    const float* Wv    = (const float*)d_in[5];
    const float* bv    = (const float*)d_in[6];
    const float* Wo    = (const float*)d_in[7];
    const float* bo    = (const float*)d_in[8];
    const float* ln1_g = (const float*)d_in[9];
    const float* ln1_b = (const float*)d_in[10];
    const float* ln2_g = (const float*)d_in[11];
    const float* ln2_b = (const float*)d_in[12];
    const float* W1    = (const float*)d_in[13];
    const float* b1    = (const float*)d_in[14];
    const float* W2    = (const float*)d_in[15];
    const float* b2    = (const float*)d_in[16];

    char* p = (char*)d_ws;
    auto alloc = [&](size_t bytes) -> void* {
        void* r = (void*)p;
        p += (bytes + 255) & ~(size_t)255;
        return r;
    };
    const size_t NW = (size_t)DDm * DDm;       // 1M weight elems
    const size_t NA = (size_t)BB * SS * DDm;   // 4M activation elems

    bf16_t* wqT   = (bf16_t*)alloc(NW * 2);    // bf16 [N,K] transposed
    bf16_t* wkT   = (bf16_t*)alloc(NW * 2);
    bf16_t* wvT   = (bf16_t*)alloc(NW * 2);
    bf16_t* woT   = (bf16_t*)alloc(NW * 2);
    bf16_t* w1T   = (bf16_t*)alloc(NW * 2);
    bf16_t* w2T   = (bf16_t*)alloc(NW * 2);
    float*  hln1_f  = (float*) alloc(NA * 4);
    bf16_t* hln1_bf = (bf16_t*)alloc(NA * 2);
    bf16_t* q_bf    = (bf16_t*)alloc(NA * 2);  // [B,H,S,DH]
    bf16_t* k_bf    = (bf16_t*)alloc(NA * 2);  // [B,H,S,DH]
    bf16_t* vT_bf   = (bf16_t*)alloc(NA * 2);  // [B,H,DH,S]  (transposed)
    bf16_t* ctx_bf  = (bf16_t*)alloc(NA * 2);  // [B,S,D]
    float*  h_f     = (float*) alloc(NA * 4);
    bf16_t* hln2_bf = (bf16_t*)alloc(NA * 2);
    bf16_t* act_bf  = (bf16_t*)alloc(NA * 2);

    const int M = BB * SS;                      // 4096
    dim3 gemm_grid(DDm / 64, M / 256);          // 16 x 16
    dim3 tgrid(DDm / 32, DDm / 32);             // weight transpose tiles

    // 1) weights fp32 [K,N] -> bf16 [N,K]
    cvt_transpose_kernel<<<tgrid, 256, 0, stream>>>(Wq, wqT, DDm, DDm);
    cvt_transpose_kernel<<<tgrid, 256, 0, stream>>>(Wk, wkT, DDm, DDm);
    cvt_transpose_kernel<<<tgrid, 256, 0, stream>>>(Wv, wvT, DDm, DDm);
    cvt_transpose_kernel<<<tgrid, 256, 0, stream>>>(Wo, woT, DDm, DDm);
    cvt_transpose_kernel<<<tgrid, 256, 0, stream>>>(W1, w1T, DDm, DDm);
    cvt_transpose_kernel<<<tgrid, 256, 0, stream>>>(W2, w2T, DDm, DDm);

    // 2) h = LN1(x)
    ln_kernel<<<M, 256, 0, stream>>>(x, ln1_g, ln1_b, hln1_f, hln1_bf);

    // 3) Q/K/V projections (Q pre-scaled by 1/sqrt(DH); V stored transposed)
    gemm_bf16_kernel<<<gemm_grid, 256, 0, stream>>>(hln1_bf, wqT, bq, M, DDm, DDm,
                                                    0, 0.125f, nullptr, nullptr, q_bf);
    gemm_bf16_kernel<<<gemm_grid, 256, 0, stream>>>(hln1_bf, wkT, bk, M, DDm, DDm,
                                                    0, 1.0f, nullptr, nullptr, k_bf);
    gemm_bf16_kernel<<<gemm_grid, 256, 0, stream>>>(hln1_bf, wvT, bv, M, DDm, DDm,
                                                    3, 1.0f, nullptr, nullptr, vT_bf);

    // 4) fused top-k attention
    attn_topk_kernel<<<BB * HHn * (SS / 16), 256, 0, stream>>>(q_bf, k_bf, vT_bf, ctx_bf);

    // 5) h = LN1(x) + ctx @ Wo + bo
    gemm_bf16_kernel<<<gemm_grid, 256, 0, stream>>>(ctx_bf, woT, bo, M, DDm, DDm,
                                                    1, 1.0f, hln1_f, h_f, nullptr);

    // 6) m = LN2(h)
    ln_kernel<<<M, 256, 0, stream>>>(h_f, ln2_g, ln2_b, nullptr, hln2_bf);

    // 7) act = GELU_exact(m @ W1 + b1)
    gemm_bf16_kernel<<<gemm_grid, 256, 0, stream>>>(hln2_bf, w1T, b1, M, DDm, DDm,
                                                    2, 1.0f, nullptr, nullptr, act_bf);

    // 8) out = h + act @ W2 + b2
    gemm_bf16_kernel<<<gemm_grid, 256, 0, stream>>>(act_bf, w2T, b2, M, DDm, DDm,
                                                    1, 1.0f, h_f, (float*)d_out, nullptr);

    (void)in_sizes; (void)n_in; (void)out_size; (void)ws_size;
}
